// MixtureOfExperts_28243704939176
// MI455X (gfx1250) — matrix operand order repaired and checked by reading, not compiled, not run on hardware
//
#include <hip/hip_runtime.h>
#include <hip/hip_bf16.h>
#include <stdint.h>

// MoE top-2 (B=2,S=2048 -> N=4096 tokens, D=512, F=1536, E=8) for gfx1250.
// bf16 WMMA 16x16x32, f32 accumulate. Sparse per-expert token bucketing
// (4x fewer FLOPs than the dense reference formulation), fused
// GEMM1 -> SwiGLU -> GEMM2 per 32-token tile, LDS-staged activations.
// Phase A is k-outer with 12 persistent accumulators so each A fragment
// (2x ds_load_b128) is amortized over 12 WMMAs; B fragments stream from L2
// as base+immediate-offset global_load_b128.
// Deterministic: no atomics anywhere; partial outputs per (token, top-k slot)
// reduced in fixed order. Workspace assumption: ws_size >= ~60 MB.

#define DD 512
#define FF 1536
#define EE 8
#define M_TILE 32
#define FC 384        // F chunk held in LDS
#define NCHUNK 4      // FC * NCHUNK == FF

typedef __attribute__((ext_vector_type(16))) __bf16 v16bf;
typedef __attribute__((ext_vector_type(8)))  float  v8f;

__device__ __forceinline__ unsigned short f2bf(float f) {
  unsigned u = __builtin_bit_cast(unsigned, f);
  unsigned r = u + 0x7fffu + ((u >> 16) & 1u);   // round-to-nearest-even
  return (unsigned short)(r >> 16);
}

// A fragment (16x32 bf16, MxK): lane<16 row m holds K=[k0..k0+15]
// with k0 = kbase + (lane>=16 ? 16 : 0).  Two 16B loads.
__device__ __forceinline__ v16bf frag_a(const unsigned short* row, int k0) {
  union { uint4 q[2]; v16bf v; } u;
  u.q[0] = *(const uint4*)(row + k0);
  u.q[1] = *(const uint4*)(row + k0 + 8);
  return u.v;
}
// B fragment (32x16 bf16, KxN): lane<16 col n holds K=[0..15], lane>=16 K=[16..31];
// p already points at the lane's 16 contiguous K values (pre-transposed weights).
__device__ __forceinline__ v16bf frag_b(const unsigned short* p) {
  union { uint4 q[2]; v16bf v; } u;
  u.q[0] = *(const uint4*)(p);
  u.q[1] = *(const uint4*)(p + 8);
  return u.v;
}
__device__ __forceinline__ v8f wmma_bf16(v16bf a, v16bf b, v8f c) {
  return __builtin_amdgcn_wmma_f32_16x16x32_bf16(false, a, false, b, (short)0, c,
                                                 false, false);
}

// ---------- K0: x fp32 -> bf16 ----------
__global__ void k_cvt_x(const float* __restrict__ x, unsigned short* __restrict__ xbf, int n) {
  int i = (blockIdx.x * blockDim.x + threadIdx.x) * 4;
  if (i >= n) return;
  float4 v = *(const float4*)(x + i);
  unsigned lo = (unsigned)f2bf(v.x) | ((unsigned)f2bf(v.y) << 16);
  unsigned hi = (unsigned)f2bf(v.z) | ((unsigned)f2bf(v.w) << 16);
  *(uint2*)(xbf + i) = make_uint2(lo, hi);
}

// ---------- K1: per-expert transpose + convert: src fp32 [R][C] -> dst bf16 [C][R] ----------
__global__ void k_transpose_bf(const float* __restrict__ src, unsigned short* __restrict__ dst,
                               int R, int C) {
  __shared__ float tile[16][17];
  int e = blockIdx.z;
  const float* s = src + (size_t)e * R * C;
  unsigned short* d = dst + (size_t)e * R * C;
  int c0 = blockIdx.x * 16, r0 = blockIdx.y * 16;
  int tx = threadIdx.x, ty = threadIdx.y;
  tile[ty][tx] = s[(size_t)(r0 + ty) * C + (c0 + tx)];
  __syncthreads();
  d[(size_t)(c0 + ty) * R + (r0 + tx)] = f2bf(tile[tx][ty]);
}

// ---------- K2: router (one wave32 per token) ----------
__global__ void k_router(const float* __restrict__ x, const float* __restrict__ wg,
                         int* __restrict__ topE, float* __restrict__ topW, int nTok) {
  int wave = threadIdx.x >> 5, lane = threadIdx.x & 31;
  int t = blockIdx.x * 8 + wave;
  if (t >= nTok) return;
  float acc[EE];
#pragma unroll
  for (int e = 0; e < EE; e++) acc[e] = 0.f;
  for (int dd = lane; dd < DD; dd += 32) {
    float xv = x[(size_t)t * DD + dd];
#pragma unroll
    for (int e = 0; e < EE; e++) acc[e] += xv * wg[dd * EE + e];
  }
#pragma unroll
  for (int e = 0; e < EE; e++)
    for (int off = 16; off; off >>= 1) acc[e] += __shfl_xor(acc[e], off, 32);
  if (lane == 0) {
    int e0 = 0; float l0 = acc[0];
    for (int e = 1; e < EE; e++) if (acc[e] > l0) { l0 = acc[e]; e0 = e; }
    int e1 = (e0 == 0) ? 1 : 0; float l1 = acc[e1];
    for (int e = 0; e < EE; e++)
      if (e != e0 && acc[e] > l1) { l1 = acc[e]; e1 = e; }
    float w1v = __expf(l1 - l0);          // l0 >= l1
    float inv = 1.f / (1.f + w1v);
    topE[t * 2] = e0; topE[t * 2 + 1] = e1;
    topW[t * 2] = inv; topW[t * 2 + 1] = w1v * inv;
  }
}

// ---------- K3: deterministic stable bucketing (single workgroup) ----------
__global__ void k_bucket(const int* __restrict__ topE, const float* __restrict__ topW,
                         int* __restrict__ counts, int* __restrict__ bTok,
                         float* __restrict__ bW, int nTok) {
  __shared__ int cnt[256][EE];
  int j = threadIdx.x;
  int per = nTok / 256;
  int t0 = j * per;
  int local[EE];
#pragma unroll
  for (int e = 0; e < EE; e++) local[e] = 0;
  for (int t = t0; t < t0 + per; t++)
    for (int k = 0; k < 2; k++) local[topE[t * 2 + k]]++;
#pragma unroll
  for (int e = 0; e < EE; e++) cnt[j][e] = local[e];
  __syncthreads();
  int start[EE];
#pragma unroll
  for (int e = 0; e < EE; e++) {
    int s = 0;
    for (int jj = 0; jj < j; jj++) s += cnt[jj][e];
    start[e] = s;
  }
  if (j == 255)
    for (int e = 0; e < EE; e++) counts[e] = start[e] + local[e];
  for (int t = t0; t < t0 + per; t++)
    for (int k = 0; k < 2; k++) {
      int e = topE[t * 2 + k];
      int pos = start[e]++;
      bTok[e * nTok + pos] = t * 2 + k;   // slot index = token*2 + k
      bW[e * nTok + pos] = topW[t * 2 + k];
    }
}

// ---------- K4: fused FFN per (expert, 32-token tile) ----------
__global__ __launch_bounds__(256) void k_moe_ffn(
    const unsigned short* __restrict__ xbf,
    const unsigned short* __restrict__ w1t,   // [E][F][D] bf16
    const unsigned short* __restrict__ w3t,   // [E][F][D] bf16
    const unsigned short* __restrict__ w2t,   // [E][D][F] bf16
    const float* __restrict__ b3,             // [E][F] fp32
    const int* __restrict__ counts,
    const int* __restrict__ bTok, const float* __restrict__ bW,
    float* __restrict__ outPartial,           // [N*2][D] fp32
    int nTok, int maxTiles) {
  __shared__ unsigned short xTile[M_TILE * DD];    // 32 KB
  __shared__ unsigned short actTile[M_TILE * FC];  // 24 KB
  __shared__ int   tokSlot[M_TILE];
  __shared__ float tokW[M_TILE];

  int e = blockIdx.x / maxTiles;
  int tile = blockIdx.x % maxTiles;
  int cnt = counts[e];
  if (tile * M_TILE >= cnt) return;
  int nRow = min(M_TILE, cnt - tile * M_TILE);

  int tid = threadIdx.x;
  if (tid < M_TILE) {
    if (tid < nRow) {
      tokSlot[tid] = bTok[e * nTok + tile * M_TILE + tid];
      tokW[tid]    = bW[e * nTok + tile * M_TILE + tid];
    } else { tokSlot[tid] = 0; tokW[tid] = 0.f; }
  }
  __syncthreads();
  { // gather X rows (bf16) into LDS once: 8 threads per row, 64 elems each
    int r = tid >> 3, seg = tid & 7;
    int tok = tokSlot[r] >> 1;
    const uint4* src = (const uint4*)(xbf + (size_t)tok * DD + seg * 64);
    uint4* dst = (uint4*)(&xTile[r * DD + seg * 64]);
#pragma unroll
    for (int q = 0; q < 8; q++) dst[q] = src[q];
  }
  __syncthreads();

  int w = tid >> 5, lane = tid & 31;
  int rt = w & 1;          // row-tile (16 tokens)
  int g  = w >> 1;         // wave group 0..3
  int hi = lane >> 4, ln = lane & 15;

  v8f acc[8];
#pragma unroll
  for (int i = 0; i < 8; i++) acc[i] = (v8f){};

  const unsigned short* arowX   = &xTile[(rt * 16 + ln) * DD];
  const unsigned short* arowAct = &actTile[(rt * 16 + ln) * FC];

  for (int c = 0; c < NCHUNK; c++) {
    // ---- Phase A: act[32][FC] = silu(X@W1) * (X@W3 + b3), bf16 into LDS ----
    // k-outer: one A fragment feeds 12 WMMAs (6 f-tiles x {W1,W3}).
    {
      int f0 = c * FC + (g * 6) * 16 + ln;       // this lane's col in f-tile 0
      const unsigned short* b1base = w1t + ((size_t)e * FF + f0) * DD;
      const unsigned short* b3base = w3t + ((size_t)e * FF + f0) * DD;
      v8f accH[6], accG[6];
#pragma unroll
      for (int i = 0; i < 6; i++) { accH[i] = (v8f){}; accG[i] = (v8f){}; }
#pragma unroll 2
      for (int kk = 0; kk < DD / 32; kk++) {
        v16bf a = frag_a(arowX, kk * 32 + hi * 16);
        int kb = kk * 32 + hi * 16;
#pragma unroll
        for (int i = 0; i < 6; i++) {
          // f-tile stride in the transposed weight layout: 16 rows * D elems
          v16bf bb1 = frag_b(b1base + (size_t)i * (16 * DD) + kb);
          v16bf bb3 = frag_b(b3base + (size_t)i * (16 * DD) + kb);
          accH[i] = wmma_bf16(a, bb1, accH[i]);
          accG[i] = wmma_bf16(a, bb3, accG[i]);
        }
      }
      // epilogue: bias + silu*gate, convert to bf16 into LDS
#pragma unroll
      for (int i = 0; i < 6; i++) {
        int ftl = g * 6 + i;
        float b3v = b3[e * FF + c * FC + ftl * 16 + ln];
#pragma unroll
        for (int j = 0; j < 8; j++) {
          float h  = accH[i][j];
          float gg = accG[i][j] + b3v;
          float s  = h / (1.f + __expf(-h));     // silu(h)
          int m = j + hi * 8;
          actTile[(rt * 16 + m) * FC + ftl * 16 + ln] = f2bf(s * gg);
        }
      }
    }
    __syncthreads();
    // ---- Phase B: accumulate out[32][512] += act_chunk @ W2[chunk rows] ----
    {
      int dcol0 = g * 128 + ln;                  // this lane's col in d-tile 0
      const unsigned short* b2base = w2t + ((size_t)e * DD + dcol0) * FF;
#pragma unroll 1
      for (int kk = 0; kk < FC / 32; kk++) {
        v16bf a = frag_a(arowAct, kk * 32 + hi * 16);
        int kb = c * FC + kk * 32 + hi * 16;
#pragma unroll
        for (int ct = 0; ct < 8; ct++) {
          // d-tile stride: 16 rows * F elems
          v16bf bb = frag_b(b2base + (size_t)ct * (16 * FF) + kb);
          acc[ct] = wmma_bf16(a, bb, acc[ct]);
        }
      }
    }
    __syncthreads();   // actTile reused next chunk
  }

  // ---- Store: fold per-token routing weight; one write per (token, slot) ----
#pragma unroll
  for (int ct = 0; ct < 8; ct++) {
    int col = g * 128 + ct * 16 + ln;
#pragma unroll
    for (int j = 0; j < 8; j++) {
      int row = rt * 16 + j + hi * 8;
      if (row < nRow) {
        outPartial[(size_t)tokSlot[row] * DD + col] = acc[ct][j] * tokW[row];
      }
    }
  }
}

// ---------- K5: out[t] = P[2t] + P[2t+1]  (fixed order -> deterministic) ----------
__global__ void k_reduce(const float* __restrict__ P, float* __restrict__ out, int n) {
  int i = (blockIdx.x * blockDim.x + threadIdx.x) * 4;
  if (i >= n) return;
  int t = i / DD, d = i % DD;
  float4 a = *(const float4*)(P + (size_t)(2 * t) * DD + d);
  float4 b = *(const float4*)(P + (size_t)(2 * t + 1) * DD + d);
  float4 r; r.x = a.x + b.x; r.y = a.y + b.y; r.z = a.z + b.z; r.w = a.w + b.w;
  *(float4*)(out + i) = r;
}

extern "C" void kernel_launch(void* const* d_in, const int* in_sizes, int n_in,
                              void* d_out, int out_size, void* d_ws, size_t ws_size,
                              hipStream_t stream) {
  const float* x  = (const float*)d_in[0];
  const float* wg = (const float*)d_in[1];
  const float* w1 = (const float*)d_in[2];
  const float* w2 = (const float*)d_in[3];
  const float* w3 = (const float*)d_in[4];
  const float* b3 = (const float*)d_in[5];
  int nTok = in_sizes[0] / DD;              // 4096

  char* ws = (char*)d_ws;
  size_t o = 0;
  auto alloc = [&](size_t bytes) { size_t r = o; o += (bytes + 255) & ~(size_t)255; return r; };
  unsigned short* xbf = (unsigned short*)(ws + alloc((size_t)nTok * DD * 2));
  unsigned short* w1t = (unsigned short*)(ws + alloc((size_t)EE * FF * DD * 2));
  unsigned short* w3t = (unsigned short*)(ws + alloc((size_t)EE * FF * DD * 2));
  unsigned short* w2t = (unsigned short*)(ws + alloc((size_t)EE * DD * FF * 2));
  int*   topE   = (int*)(ws + alloc((size_t)nTok * 2 * 4));
  float* topW   = (float*)(ws + alloc((size_t)nTok * 2 * 4));
  int*   counts = (int*)(ws + alloc(EE * 4));
  int*   bTok   = (int*)(ws + alloc((size_t)EE * nTok * 4));
  float* bW     = (float*)(ws + alloc((size_t)EE * nTok * 4));
  float* outP   = (float*)(ws + alloc((size_t)nTok * 2 * DD * 4));
  (void)ws_size; (void)n_in; (void)out_size;

  k_cvt_x<<<(nTok * DD / 4 + 255) / 256, 256, 0, stream>>>(x, xbf, nTok * DD);
  k_transpose_bf<<<dim3(FF / 16, DD / 16, EE), dim3(16, 16), 0, stream>>>(w1, w1t, DD, FF);
  k_transpose_bf<<<dim3(FF / 16, DD / 16, EE), dim3(16, 16), 0, stream>>>(w3, w3t, DD, FF);
  k_transpose_bf<<<dim3(DD / 16, FF / 16, EE), dim3(16, 16), 0, stream>>>(w2, w2t, FF, DD);
  k_router<<<(nTok + 7) / 8, 256, 0, stream>>>(x, wg, topE, topW, nTok);
  k_bucket<<<1, 256, 0, stream>>>(topE, topW, counts, bTok, bW, nTok);
  int maxTiles = (nTok + M_TILE - 1) / M_TILE;
  k_moe_ffn<<<EE * maxTiles, 256, 0, stream>>>(xbf, w1t, w3t, w2t, b3, counts,
                                               bTok, bW, outP, nTok, maxTiles);
  k_reduce<<<(nTok * DD / 4 + 255) / 256, 256, 0, stream>>>(outP, (float*)d_out, nTok * DD);
}
